// OffGridPatchEmbedGatherConv2D_1297080123693
// MI455X (gfx1250) — matrix operand order repaired and checked by reading, not compiled, not run on hardware
//
#include <hip/hip_runtime.h>

typedef float  v8f  __attribute__((ext_vector_type(8)));
typedef __bf16 v8bf  __attribute__((ext_vector_type(8)));
typedef __bf16 v16bf __attribute__((ext_vector_type(16)));

#define B_  64
#define C_  3
#define H_  384
#define W_  384
#define P_  16
#define E_  768
#define N_  576              // (H/P)*(W/P)
#define K_  768              // C*P*P
#define M_  (B_ * N_)        // 36864 tokens

#define TILE_M 64
#define TILE_E 64
#define TILE_K 64
#define NTHREADS 128         // 4 wave32 waves

#define LSTRIDE 72           // bf16 elems per LDS row (144B): rows land 36 banks apart ->
                             // 16 b128 lane-reads tile all 64 banks exactly once (conflict-free)

// Split one fp32 into two bf16 limbs: f ~= hi + lo  (RNE per limb, ~16 mantissa bits total)
__device__ __forceinline__ void split_bf16(float f, __bf16& h, __bf16& l) {
    h = (__bf16)f;
    l = (__bf16)(f - (float)h);
}

// Concatenate two contiguous 8x-bf16 LDS loads into a WMMA v16bf fragment.
__device__ __forceinline__ v16bf ld_frag(const __bf16* p0, const __bf16* p1) {
    v8bf a = *(const v8bf*)p0;
    v8bf b = *(const v8bf*)p1;
    return __builtin_shufflevector(a, b, 0,1,2,3,4,5,6,7,8,9,10,11,12,13,14,15);
}

__global__ __launch_bounds__(NTHREADS)
void patch_embed_wmma_bf16x2(const float* __restrict__ x,
                             const int*   __restrict__ ys,
                             const int*   __restrict__ xs,
                             const float* __restrict__ w,
                             const float* __restrict__ bias,
                             float*       __restrict__ out)
{
    // hi/lo bf16 planes; A as [tok][k], B as [e][k] (B-frag wants contiguous K at fixed e)
    __shared__ __align__(16) __bf16 Ah[TILE_M * LSTRIDE];
    __shared__ __align__(16) __bf16 Al[TILE_M * LSTRIDE];
    __shared__ __align__(16) __bf16 Bh[TILE_E * LSTRIDE];
    __shared__ __align__(16) __bf16 Bl[TILE_E * LSTRIDE];
    __shared__ int sYs[TILE_M];
    __shared__ int sXs[TILE_M];

    const int t    = threadIdx.x;
    const int lane = t & 31;
    const int wave = t >> 5;
    const int lm   = lane & 15;
    const int half = lane >> 4;          // 0: lanes 0-15, 1: lanes 16-31

    const int m0 = blockIdx.x * TILE_M;  // token tile base
    const int e0 = blockIdx.y * TILE_E;  // embed tile base

    if (t < TILE_M) {
        sYs[t] = ys[m0 + t];
        sXs[t] = xs[m0 + t];
    }
    __syncthreads();

    v8f acc[4];
    #pragma unroll
    for (int s = 0; s < 4; ++s) acc[s] = (v8f){0.f,0.f,0.f,0.f,0.f,0.f,0.f,0.f};

    #pragma unroll 1
    for (int ci = 0; ci < K_ / TILE_K; ++ci) {
        const int kc0 = ci * TILE_K;
        const int c   = kc0 >> 8;            // channel (256 k per channel; chunk never crosses)
        const int py0 = (kc0 & 255) >> 4;    // first patch row of this chunk

        // ---- stage A: gather patch rows, split fp32 -> bf16 hi/lo planes ----
        #pragma unroll
        for (int j = 0; j < 2; ++j) {
            const int R    = t * 2 + j;      // 0..255  (64 tokens x 4 patch rows)
            const int tok  = R >> 2;
            const int pr   = R & 3;
            const int m    = m0 + tok;
            const int bimg = m / N_;
            const float* src = x + (((size_t)bimg * C_ + c) * H_ + (size_t)(sYs[tok] + py0 + pr)) * W_
                                 + sXs[tok];
            const int base = tok * LSTRIDE + pr * 16;
            #pragma unroll
            for (int g = 0; g < 2; ++g) {
                v8bf hv, lv;
                #pragma unroll
                for (int q = 0; q < 8; ++q) {
                    __bf16 h, l;
                    split_bf16(src[g * 8 + q], h, l);
                    hv[q] = h; lv[q] = l;
                }
                *(v8bf*)&Ah[base + g * 8] = hv;
                *(v8bf*)&Al[base + g * 8] = lv;
            }
        }

        // ---- stage B: copy weight rows (no transpose), split hi/lo ----
        {
            const int e  = t >> 1;
            const int hf = t & 1;
            const float* src = w + (size_t)(e0 + e) * K_ + kc0 + hf * 32;
            const int base = e * LSTRIDE + hf * 32;
            #pragma unroll
            for (int g = 0; g < 4; ++g) {
                v8bf hv, lv;
                #pragma unroll
                for (int q = 0; q < 8; ++q) {
                    __bf16 h, l;
                    split_bf16(src[g * 8 + q], h, l);
                    hv[q] = h; lv[q] = l;
                }
                *(v8bf*)&Bh[base + g * 8] = hv;
                *(v8bf*)&Bl[base + g * 8] = lv;
            }
            if (ci + 1 < K_ / TILE_K)        // warm L2 for next weight chunk (global_prefetch_b8)
                __builtin_prefetch(src + TILE_K, 0, 0);
        }

        __syncthreads();

        // ---- compute: 2 k-steps of K=32, 3 bf16 WMMAs per 16x16 subtile (hh, hl, lh) ----
        const int aRow = wave * 16 + lm;
        const int aOff = half * 16;          // A frag: lanes<16 K=0..15; lanes>=16 K=16..31
        const int bOff = half * 16;          // B frag: lanes<16 K=0..15; lanes>=16 K=16..31
        #pragma unroll
        for (int k0 = 0; k0 < TILE_K; k0 += 32) {
            const __bf16* ahp = &Ah[aRow * LSTRIDE + k0 + aOff];
            const __bf16* alp = &Al[aRow * LSTRIDE + k0 + aOff];
            v16bf ah = ld_frag(ahp, ahp + 8);
            v16bf al = ld_frag(alp, alp + 8);
            #pragma unroll
            for (int s = 0; s < 4; ++s) {
                const __bf16* bhp = &Bh[(s * 16 + lm) * LSTRIDE + k0 + bOff];
                const __bf16* blp = &Bl[(s * 16 + lm) * LSTRIDE + k0 + bOff];
                v16bf bh = ld_frag(bhp, bhp + 8);
                v16bf bl = ld_frag(blp, blp + 8);
                acc[s] = __builtin_amdgcn_wmma_f32_16x16x32_bf16(
                    false, ah, false, bh, (short)0, acc[s], false, false);
                acc[s] = __builtin_amdgcn_wmma_f32_16x16x32_bf16(
                    false, ah, false, bl, (short)0, acc[s], false, false);
                acc[s] = __builtin_amdgcn_wmma_f32_16x16x32_bf16(
                    false, al, false, bh, (short)0, acc[s], false, false);
            }
        }

        __syncthreads();
    }

    // ---- epilogue: add bias, store (C/D: VGPR r -> M=r lanes 0-15 / M=r+8 lanes 16-31) ----
    #pragma unroll
    for (int s = 0; s < 4; ++s) {
        const int   e  = e0 + s * 16 + lm;
        const float bv = bias[e];
        #pragma unroll
        for (int r = 0; r < 8; ++r) {
            const int m = m0 + wave * 16 + r + half * 8;
            out[(size_t)m * E_ + e] = acc[s][r] + bv;
        }
    }
}

__global__ __launch_bounds__(256)
void write_positions(const int* __restrict__ ys, const int* __restrict__ xs,
                     int* __restrict__ outPos)
{
    const int i = blockIdx.x * blockDim.x + threadIdx.x;
    if (i < M_) {
        outPos[2 * i + 0] = ys[i];
        outPos[2 * i + 1] = xs[i];
    }
}

extern "C" void kernel_launch(void* const* d_in, const int* in_sizes, int n_in,
                              void* d_out, int out_size, void* d_ws, size_t ws_size,
                              hipStream_t stream)
{
    const float* x  = (const float*)d_in[0];
    const int*   ys = (const int*)  d_in[1];
    const int*   xs = (const int*)  d_in[2];
    const float* w  = (const float*)d_in[3];
    const float* b  = (const float*)d_in[4];
    float* out = (float*)d_out;

    dim3 grid(M_ / TILE_M, E_ / TILE_E);   // 576 x 12
    patch_embed_wmma_bf16x2<<<grid, NTHREADS, 0, stream>>>(x, ys, xs, w, b, out);

    int* outPos = reinterpret_cast<int*>(out + (size_t)M_ * E_);
    write_positions<<<(M_ + 255) / 256, 256, 0, stream>>>(ys, xs, outPos);
}
